// OptimizedCustomLSTM3_10471130268443
// MI455X (gfx1250) — compile-verified, matrix operand order checked by reading, not run on hardware
//
#include <hip/hip_runtime.h>
#include <hip/hip_bf16.h>
#include <stdint.h>

// ---------------------------------------------------------------------------
// LSTM for MI455X (gfx1250): bf16 WMMA for both GEMMs, async global->LDS
// staging (h panel per step, Wh strip once per kernel), two-register-set
// software pipeline (no rotation movs / hazard nops), persistent recurrent
// kernel with grid-wide barrier, LDS-reduced K-split, early xg prefetch.
// ---------------------------------------------------------------------------

typedef __attribute__((ext_vector_type(16))) __bf16 v16bf;
typedef __attribute__((ext_vector_type(8)))  float  v8f;

#define B_    64
#define S_    512
#define I_    1024
#define H_    1024
#define G_    4096              // 4*H
#define ROWS_ (B_ * S_)         // 32768

// ---- workspace carving (bytes) --------------------------------------------
#define OFF_XG      0ull
#define SZ_XG       ((unsigned long long)ROWS_ * G_ * 4ull)   // 512 MB fp32
#define OFF_XPACK   (OFF_XG + SZ_XG)
#define SZ_XPACK    ((unsigned long long)ROWS_ * I_ * 2ull)   // 64 MB bf16
#define OFF_WXPACK  (OFF_XPACK + SZ_XPACK)
#define SZ_WXPACK   ((unsigned long long)I_ * G_ * 2ull)      // 8 MB
#define OFF_WHPACK  (OFF_WXPACK + SZ_WXPACK)
#define SZ_WHPACK   ((unsigned long long)H_ * G_ * 2ull)      // 8 MB
#define OFF_HPACK   (OFF_WHPACK + SZ_WHPACK)
#define SZ_HPACK    (2ull * B_ * H_ * 2ull)                   // 256 KB ping-pong
#define OFF_CWS     (OFF_HPACK + SZ_HPACK)
#define SZ_CWS      ((unsigned long long)B_ * H_ * 4ull)      // 256 KB
#define OFF_BAR     (OFF_CWS + SZ_CWS)                        // 2 x u32

// ---- CDNA5 async global->LDS copy (ASYNCcnt path) -------------------------
__device__ __forceinline__ void async_g2l_b128(uint32_t lds_off, const void* g) {
    // VDST = per-lane LDS byte offset, VADDR = 64-bit global address
    asm volatile("global_load_async_to_lds_b128 %0, %1, off"
                 :: "v"(lds_off), "v"((uint64_t)(uintptr_t)g)
                 : "memory");
}
__device__ __forceinline__ void wait_async0() {
    asm volatile("s_wait_asynccnt 0x0" ::: "memory");
}

// ---- helpers --------------------------------------------------------------
__device__ __forceinline__ uint16_t f2bf(float f) {
    unsigned u = __float_as_uint(f);
    u += 0x7fffu + ((u >> 16) & 1u);     // round-to-nearest-even
    return (uint16_t)(u >> 16);
}

// Per-lane placement for the 16-bit WMMA A-matrix 16x32 (M x K) layout.
__device__ __forceinline__ void a_lane_elem(int m, int k, int& lane, int& e) {
    if      (k < 8)  { lane = m;      e = k;      }
    else if (k < 16) { lane = m + 16; e = k - 8;  }
    else if (k < 24) { lane = m;      e = k - 8;  }
    else             { lane = m + 16; e = k - 16; }
}

__device__ __forceinline__ float sigmoidf_(float x) {
    return 1.f / (1.f + __expf(-x));
}

// Grid-wide barrier over NB co-resident blocks (counter+generation in ws).
__device__ __forceinline__ void gridbar(unsigned* bar, unsigned nb) {
    __syncthreads();
    __threadfence();
    if (threadIdx.x == 0) {
        volatile unsigned* vgen = bar + 1;
        unsigned g = *vgen;
        if (atomicAdd(bar, 1u) == nb - 1u) {
            atomicExch(bar, 0u);
            __threadfence();
            atomicAdd(bar + 1, 1u);
        } else {
            while (*vgen == g) { __builtin_amdgcn_s_sleep(1); }
        }
    }
    __threadfence();
    __syncthreads();
}

#define WMMA_BF16(A, Bm, C) \
    __builtin_amdgcn_wmma_f32_16x16x32_bf16(false, (A), false, (Bm), (short)0, (C), false, false)

// ---- pack x (fp32 row-major -> bf16 WMMA-A tile layout) -------------------
__global__ void k_pack_x(const float* __restrict__ x, uint16_t* __restrict__ xp) {
    long idx = (long)blockIdx.x * blockDim.x + threadIdx.x;   // < ROWS_*I_
    int row = (int)(idx >> 10);
    int col = (int)(idx & 1023);
    int mtile = row >> 4, m = row & 15;
    int ktile = col >> 5, k = col & 31;
    int lane, e; a_lane_elem(m, k, lane, e);
    long dst = ((((long)mtile * (I_ / 32) + ktile) * 32 + lane) << 4) + e;
    xp[dst] = f2bf(x[idx]);
}

// ---- pack W (fp32 -> bf16 WMMA-B tile layout; split Wx / Wh) --------------
__global__ void k_pack_w(const float* __restrict__ W,
                         uint16_t* __restrict__ wxp, uint16_t* __restrict__ whp) {
    long idx = (long)blockIdx.x * blockDim.x + threadIdx.x;   // < 2048*4096
    int kg = (int)(idx >> 12);
    int n  = (int)(idx & 4095);
    uint16_t v = f2bf(W[idx]);
    uint16_t* dstbuf;
    int kk;
    if (kg < I_) { dstbuf = wxp; kk = kg; } else { dstbuf = whp; kk = kg - I_; }
    int ntile = n >> 4, n16 = n & 15;
    int ktile = kk >> 5, k = kk & 31;
    int lane, e; a_lane_elem(n16, k, lane, e);   // symmetric B layout (lane=N col)
    long dst = (((long)ntile * 32 + ktile) * 512) + lane * 16 + e;
    dstbuf[dst] = v;
}

// ---- big GEMM: xg = x @ Wx + b  (32768x1024 * 1024x4096) ------------------
// Block: 256 threads = 8 waves; 64-row x 128-col block. 128 KB A panel async-
// copied to LDS once, shared by all 8 waves. Two-register-set pipeline: loads
// for tile kt+1 are issued between the WMMA groups of tile kt (no rotation
// movs, nonzero wait thresholds).
__global__ void __launch_bounds__(256, 2)
k_gemm_xg(const uint16_t* __restrict__ xp, const uint16_t* __restrict__ wxp,
          const float* __restrict__ bias, float* __restrict__ xg) {
    __shared__ uint16_t lds_a[B_ * I_];          // 128 KB A panel
    int nblock = blockIdx.x & 31;     // 32 column blocks of 128
    int mblock = blockIdx.x >> 5;     // 512 row blocks of 64
    int tid  = threadIdx.x;
    int wave = tid >> 5;
    int lane = tid & 31;

    // async copy A panel: 131072 bytes, 512 B per thread (4 x b128)
    {
        const char* gsrc = (const char*)(xp + (long)mblock * 4 * 32 * 512);
        uint32_t lbase = (uint32_t)(uintptr_t)lds_a + tid * 512u;
        #pragma unroll
        for (int i = 0; i < 4; ++i)
            async_g2l_b128(lbase + i * 128u, gsrc + tid * 512 + i * 128);
        wait_async0();
    }
    __syncthreads();

    int ntile = nblock * 8 + wave;
    const v16bf* bsrc = (const v16bf*)(wxp + (long)ntile * 32 * 512);
    const v16bf* aL = (const v16bf*)lds_a;

    v8f acc0 = {}, acc1 = {}, acc2 = {}, acc3 = {};
    v16bf bA  = bsrc[lane];
    v16bf aA0 = aL[(0 * 32 + 0) * 32 + lane];
    v16bf aA1 = aL[(1 * 32 + 0) * 32 + lane];
    v16bf aA2 = aL[(2 * 32 + 0) * 32 + lane];
    v16bf aA3 = aL[(3 * 32 + 0) * 32 + lane];
    #pragma unroll 1
    for (int kt = 0; kt < 32; kt += 2) {
        // stage set B for kt+1 (trip count even -> always in range)
        v16bf bB  = bsrc[(kt + 1) * 32 + lane];
        v16bf aB0 = aL[(0 * 32 + kt + 1) * 32 + lane];
        v16bf aB1 = aL[(1 * 32 + kt + 1) * 32 + lane];
        v16bf aB2 = aL[(2 * 32 + kt + 1) * 32 + lane];
        v16bf aB3 = aL[(3 * 32 + kt + 1) * 32 + lane];
        acc0 = WMMA_BF16(aA0, bA, acc0);
        acc1 = WMMA_BF16(aA1, bA, acc1);
        acc2 = WMMA_BF16(aA2, bA, acc2);
        acc3 = WMMA_BF16(aA3, bA, acc3);
        // stage set A for kt+2 (clamped dummy reload on the last iteration)
        int k2 = (kt + 2 < 32) ? kt + 2 : 0;
        bA  = bsrc[k2 * 32 + lane];
        aA0 = aL[(0 * 32 + k2) * 32 + lane];
        aA1 = aL[(1 * 32 + k2) * 32 + lane];
        aA2 = aL[(2 * 32 + k2) * 32 + lane];
        aA3 = aL[(3 * 32 + k2) * 32 + lane];
        acc0 = WMMA_BF16(aB0, bB, acc0);
        acc1 = WMMA_BF16(aB1, bB, acc1);
        acc2 = WMMA_BF16(aB2, bB, acc2);
        acc3 = WMMA_BF16(aB3, bB, acc3);
    }

    // C layout: VGPR r, lanes 0-15 -> M=r (N=lane); lanes 16-31 -> M=r+8.
    int nc = ntile * 16 + (lane & 15);
    float bv = bias[nc];
    int rofs = (lane < 16) ? 0 : 8;
    #pragma unroll
    for (int r = 0; r < 8; ++r) {
        xg[(long)(mblock * 64 +  0 + rofs + r) * G_ + nc] = acc0[r] + bv;
        xg[(long)(mblock * 64 + 16 + rofs + r) * G_ + nc] = acc1[r] + bv;
        xg[(long)(mblock * 64 + 32 + rofs + r) * G_ + nc] = acc2[r] + bv;
        xg[(long)(mblock * 64 + 48 + rofs + r) * G_ + nc] = acc3[r] + bv;
    }
}

// ---- persistent recurrent kernel ------------------------------------------
// 64 blocks x 256 threads (8 waves). Block b owns hidden cols [16b, 16b+16).
// Wave w: gate g = w&3, K-half kh = w>>2 (64 WMMAs per wave, 2 waves/SIMD).
// LDS (288 KB of the 320 KB WGP budget): 128 KB Wh strip (loaded ONCE, B is
// step-invariant), 128 KB h panel (async-copied per step), 32 KB partials.
// Inner loop is pure LDS + WMMA; only the h ping-pong, xg gather and outputs
// touch global memory per step.
__global__ void __launch_bounds__(256, 1)
k_lstm(const uint16_t* __restrict__ whp, const float* __restrict__ xg,
       float* __restrict__ out, float* __restrict__ ht, float* __restrict__ ct,
       uint16_t* __restrict__ hpack, float* __restrict__ cws,
       unsigned* __restrict__ bar) {
    __shared__ uint16_t lds_b[4 * 32 * 512];     // 128 KB Wh strips (4 gates)
    __shared__ uint16_t lds_a[B_ * H_];          // 128 KB h panel (packed)
    __shared__ float lds_g[2][4][B_][16];        // 32 KB gate partials
    const int tid  = threadIdx.x;
    const int wave = tid >> 5, lane = tid & 31;
    const int blk  = blockIdx.x;                 // 0..63
    const int j0   = blk * 16;

    // async copy this block's 4 Wh gate strips (32 KB each) into lds_b.
    // logical offset o in [0,128K); strip g' = o>>15, byte-within = o&32767.
    {
        int o = tid * 512;
        int gg = o >> 15;
        int within = o & 32767;
        const char* gsrc = (const char*)whp + ((long)(gg * 64 + blk) * 32768) + within;
        uint32_t lbase = (uint32_t)(uintptr_t)lds_b + (uint32_t)o;
        #pragma unroll
        for (int i = 0; i < 4; ++i)
            async_g2l_b128(lbase + i * 128u, gsrc + i * 128);
    }
    // init c = 0 (own columns) and h_pack ping buffer 0 = 0
    #pragma unroll
    for (int u = 0; u < 4; ++u) {
        int idx = tid + u * 256;                 // 0..1023
        cws[(idx >> 4) * H_ + j0 + (idx & 15)] = 0.f;
        hpack[blk * 1024 + idx] = 0;
    }
    wait_async0();
    gridbar(bar, 64);

    const int g  = wave & 3;                     // gate: 0=i 1=f 2=g 3=o
    const int kh = wave >> 2;                    // K half: ktiles [kh*16, kh*16+16)
    const v16bf* bL = (const v16bf*)(lds_b + (long)g * 32 * 512);
    const v16bf* aL = (const v16bf*)lds_a;
    const int nc = g * 1024 + j0 + (lane & 15);
    const int rofs = (lane < 16) ? 0 : 8;

    for (int t = 0; t < S_; ++t) {
        // async copy this step's h panel (128 KB) into LDS, 512 B per thread
        {
            const char* gsrc = (const char*)(hpack + (long)(t & 1) * (B_ * H_));
            uint32_t lbase = (uint32_t)(uintptr_t)lds_a + tid * 512u;
            #pragma unroll
            for (int i = 0; i < 4; ++i)
                async_g2l_b128(lbase + i * 128u, gsrc + tid * 512 + i * 128);
        }
        // warm this step's xg lines while the async copy + GEMM run
        if (kh == 0) {
            #pragma unroll
            for (int r = 0; r < 8; ++r) {
                int mm = rofs + r;
                __builtin_prefetch(&xg[((long)(mm +  0) * S_ + t) * G_ + nc], 0, 1);
                __builtin_prefetch(&xg[((long)(mm + 16) * S_ + t) * G_ + nc], 0, 1);
                __builtin_prefetch(&xg[((long)(mm + 32) * S_ + t) * G_ + nc], 0, 1);
                __builtin_prefetch(&xg[((long)(mm + 48) * S_ + t) * G_ + nc], 0, 1);
            }
        }
        wait_async0();
        __syncthreads();

        v8f acc0 = {}, acc1 = {}, acc2 = {}, acc3 = {};
        const int kt0 = kh * 16, kt1 = kt0 + 16;
        v16bf bA  = bL[kt0 * 32 + lane];
        v16bf aA0 = aL[(0 * 32 + kt0) * 32 + lane];
        v16bf aA1 = aL[(1 * 32 + kt0) * 32 + lane];
        v16bf aA2 = aL[(2 * 32 + kt0) * 32 + lane];
        v16bf aA3 = aL[(3 * 32 + kt0) * 32 + lane];
        #pragma unroll 1
        for (int kt = kt0; kt < kt1; kt += 2) {
            v16bf bB  = bL[(kt + 1) * 32 + lane];
            v16bf aB0 = aL[(0 * 32 + kt + 1) * 32 + lane];
            v16bf aB1 = aL[(1 * 32 + kt + 1) * 32 + lane];
            v16bf aB2 = aL[(2 * 32 + kt + 1) * 32 + lane];
            v16bf aB3 = aL[(3 * 32 + kt + 1) * 32 + lane];
            acc0 = WMMA_BF16(aA0, bA, acc0);
            acc1 = WMMA_BF16(aA1, bA, acc1);
            acc2 = WMMA_BF16(aA2, bA, acc2);
            acc3 = WMMA_BF16(aA3, bA, acc3);
            int k2 = (kt + 2 < kt1) ? kt + 2 : kt0;
            bA  = bL[k2 * 32 + lane];
            aA0 = aL[(0 * 32 + k2) * 32 + lane];
            aA1 = aL[(1 * 32 + k2) * 32 + lane];
            aA2 = aL[(2 * 32 + k2) * 32 + lane];
            aA3 = aL[(3 * 32 + k2) * 32 + lane];
            acc0 = WMMA_BF16(aB0, bB, acc0);
            acc1 = WMMA_BF16(aB1, bB, acc1);
            acc2 = WMMA_BF16(aB2, bB, acc2);
            acc3 = WMMA_BF16(aB3, bB, acc3);
        }

        // partials -> LDS (kh==0 wave folds in xg, which already contains bias)
        int jl = lane & 15;
        #pragma unroll
        for (int r = 0; r < 8; ++r) {
            int m0 =  0 + rofs + r, m1 = 16 + rofs + r;
            int m2 = 32 + rofs + r, m3 = 48 + rofs + r;
            float x0 = 0.f, x1 = 0.f, x2 = 0.f, x3 = 0.f;
            if (kh == 0) {
                x0 = xg[((long)m0 * S_ + t) * G_ + nc];
                x1 = xg[((long)m1 * S_ + t) * G_ + nc];
                x2 = xg[((long)m2 * S_ + t) * G_ + nc];
                x3 = xg[((long)m3 * S_ + t) * G_ + nc];
            }
            lds_g[kh][g][m0][jl] = acc0[r] + x0;
            lds_g[kh][g][m1][jl] = acc1[r] + x1;
            lds_g[kh][g][m2][jl] = acc2[r] + x2;
            lds_g[kh][g][m3][jl] = acc3[r] + x3;
        }
        __syncthreads();

        // elementwise LSTM cell update for 64 rows x 16 cols
        uint16_t* hdst = hpack + (long)((t + 1) & 1) * (B_ * H_);
        #pragma unroll
        for (int u = 0; u < 4; ++u) {
            int idx = tid + u * 256;
            int m = idx >> 4, j = idx & 15;
            float iv = lds_g[0][0][m][j] + lds_g[1][0][m][j];
            float fv = lds_g[0][1][m][j] + lds_g[1][1][m][j];
            float gv = lds_g[0][2][m][j] + lds_g[1][2][m][j];
            float ov = lds_g[0][3][m][j] + lds_g[1][3][m][j];
            float cold = cws[m * H_ + j0 + j];
            float cn = sigmoidf_(fv) * cold + sigmoidf_(iv) * tanhf(gv);
            float hn = sigmoidf_(ov) * tanhf(cn);
            cws[m * H_ + j0 + j] = cn;
            out[((long)m * S_ + t) * H_ + j0 + j] = hn;
            if (t == S_ - 1) {
                ht[m * H_ + j0 + j] = hn;
                ct[m * H_ + j0 + j] = cn;
            }
            // repack h -> bf16 A-layout for next step
            int kidx = j0 + j;
            int ktile = kidx >> 5, k = kidx & 31;
            int la, e; a_lane_elem(m & 15, k, la, e);
            hdst[(((long)(m >> 4) * 32 + ktile) * 512) + la * 16 + e] = f2bf(hn);
        }
        gridbar(bar, 64);
    }
}

__global__ void k_init_bar(unsigned* bar) {
    if (threadIdx.x < 2) bar[threadIdx.x] = 0u;
}

// ---------------------------------------------------------------------------
extern "C" void kernel_launch(void* const* d_in, const int* in_sizes, int n_in,
                              void* d_out, int out_size, void* d_ws, size_t ws_size,
                              hipStream_t stream) {
    const float* x = (const float*)d_in[0];   // (B,S,I)
    const float* W = (const float*)d_in[1];   // (I+H, 4H)
    const float* b = (const float*)d_in[2];   // (4H,)

    float* out = (float*)d_out;               // outputs (B,S,H) ++ h_t ++ c_t
    float* ht  = out + (long)B_ * S_ * H_;
    float* ct  = ht + (long)B_ * H_;

    char* ws = (char*)d_ws;
    float*    xg    = (float*)   (ws + OFF_XG);
    uint16_t* xp    = (uint16_t*)(ws + OFF_XPACK);
    uint16_t* wxp   = (uint16_t*)(ws + OFF_WXPACK);
    uint16_t* whp   = (uint16_t*)(ws + OFF_WHPACK);
    uint16_t* hpack = (uint16_t*)(ws + OFF_HPACK);
    float*    cws   = (float*)   (ws + OFF_CWS);
    unsigned* bar   = (unsigned*)(ws + OFF_BAR);

    k_init_bar<<<1, 32, 0, stream>>>(bar);
    k_pack_x<<<(ROWS_ * I_) / 256, 256, 0, stream>>>(x, xp);
    k_pack_w<<<((I_ + H_) * G_) / 256, 256, 0, stream>>>(W, wxp, whp);
    k_gemm_xg<<<(ROWS_ / 64) * (G_ / 128), 256, 0, stream>>>(xp, wxp, b, xg);
    k_lstm<<<64, 256, 0, stream>>>(whp, xg, out, ht, ct, hpack, cws, bar);
}
